// DualStreamLSTM_53566832115848
// MI455X (gfx1250) — matrix operand order repaired and checked by reading, not compiled, not run on hardware
//
#include <hip/hip_runtime.h>
#include <hip/hip_bf16.h>

typedef __attribute__((ext_vector_type(16))) _Float16 v16h;
typedef __attribute__((ext_vector_type(8)))  float    v8f;

#define B_TOT   2048
#define SEQ     512
#define DIN     5
#define HID     64
#define EDIM    32
#define NEMB    64
#define CCONV   32
#define BT      32        // batch rows per workgroup
#define NTHREADS 256

// ---- LDS layout (bytes) ----
#define OFF_WT0C   0          // [256][96]  f16  49152
#define OFF_WT1C   49152      // [256][128] f16  65536
#define OFF_WT0V   114688     // [256][96]  f16  49152
#define OFF_WT1V   163840     // [256][128] f16  65536
#define OFF_A0C    229376     // [32][96]   f16   6144
#define OFF_A1C    235520     // [32][128]  f16   8192
#define OFF_A0V    243712     // [32][96]   f16   6144
#define OFF_A1V    249856     // [32][128]  f16   8192
#define OFF_B0C    258048     // [256] f32        1024
#define OFF_B1C    259072
#define OFF_B0V    260096
#define OFF_B1V    261120
#define OFF_CONVW  262144     // [32][16] f32     2048
#define OFF_VQP    264192     // [32][32] f32     4096
#define SMEM_BYTES 268288

union Half16 { v16h v; uint4 q[2]; _Float16 h[16]; };

__device__ __forceinline__ float fsig(float x)  { return 1.0f / (1.0f + __expf(-x)); }
__device__ __forceinline__ float ftanh(float x) { return 2.0f / (1.0f + __expf(-2.0f * x)) - 1.0f; }

// One LSTM step for this wave's (mb, jt) slice.
// A  : [BT][KDIM] f16 LDS, input = [x | h_prev]
// WT : [256][KDIM] f16 LDS, row n = gate output unit, cols = stacked [Wih | Whh]
// Each wave computes gates i,f,g,o for rows mb*16..+15, units jt*16..+15,
// so the cell update is fully register-resident per lane.
template <int KDIM>
__device__ __forceinline__ void lstm_step(const _Float16* __restrict__ A,
                                          const _Float16* __restrict__ WT,
                                          const float* __restrict__ bias,
                                          int lane, int mb, int jt,
                                          v8f& cst, float* __restrict__ hreg)
{
    constexpr int NK = KDIM / 32;
    const int mrow = mb * 16 + (lane & 15);
    const int ka   = (lane & 16) ? 8 : 0;    // A: lanes 16-31 hold K 8-15 / 24-31
    Half16 af[NK];
#pragma unroll
    for (int kc = 0; kc < NK; ++kc) {
        const _Float16* ap = A + mrow * KDIM + kc * 32 + ka;
        af[kc].q[0] = *(const uint4*)(ap);        // K +0..7
        af[kc].q[1] = *(const uint4*)(ap + 16);   // K +16..23
    }
    const int ncol = lane & 15;
    const int kb   = (lane & 16) ? 8 : 0;    // B: lanes 16-31 hold K 8-15 / 24-31
    v8f acc[4];
#pragma unroll
    for (int g = 0; g < 4; ++g) {
        const int n = g * 64 + jt * 16 + ncol;
        const float bv = bias[n];
        v8f a = {bv, bv, bv, bv, bv, bv, bv, bv};
#pragma unroll
        for (int kc = 0; kc < NK; ++kc) {
            const _Float16* bp = WT + n * KDIM + kc * 32 + kb;
            Half16 bf;
            bf.q[0] = *(const uint4*)(bp);
            bf.q[1] = *(const uint4*)(bp + 16);
            a = __builtin_amdgcn_wmma_f32_16x16x32_f16(
                    false, af[kc].v, false, bf.v, (short)0, a, false, false);
        }
        acc[g] = a;
    }
    // LSTM cell, all four gates for same (b, j) are in-register here.
#pragma unroll
    for (int r = 0; r < 8; ++r) {
        float ii = fsig(acc[0][r]);
        float ff = fsig(acc[1][r]);
        float gg = ftanh(acc[2][r]);
        float oo = fsig(acc[3][r]);
        float cn = ff * cst[r] + ii * gg;
        cst[r] = cn;
        hreg[r] = oo * ftanh(cn);
    }
}

__device__ __forceinline__ void write_h(_Float16* __restrict__ dst, int kd, int coff,
                                        int lane, int mb, int jt,
                                        const float* __restrict__ hreg)
{
    const int j  = jt * 16 + (lane & 15);
    const int rb = mb * 16 + ((lane & 16) ? 8 : 0);
#pragma unroll
    for (int r = 0; r < 8; ++r)
        dst[(rb + r) * kd + coff + j] = (_Float16)hreg[r];
}

__global__ __launch_bounds__(NTHREADS)
void dual_lstm_fused(const float* __restrict__ x,
                     const float* __restrict__ conv_w, const float* __restrict__ conv_b,
                     const float* __restrict__ bn_g,  const float* __restrict__ bn_b,
                     const float* __restrict__ bn_m,  const float* __restrict__ bn_v,
                     const float* __restrict__ w0c_ih, const float* __restrict__ w0c_hh,
                     const float* __restrict__ b0c_ih, const float* __restrict__ b0c_hh,
                     const float* __restrict__ w1c_ih, const float* __restrict__ w1c_hh,
                     const float* __restrict__ b1c_ih, const float* __restrict__ b1c_hh,
                     const float* __restrict__ w0v_ih, const float* __restrict__ w0v_hh,
                     const float* __restrict__ b0v_ih, const float* __restrict__ b0v_hh,
                     const float* __restrict__ w1v_ih, const float* __restrict__ w1v_hh,
                     const float* __restrict__ b1v_ih, const float* __restrict__ b1v_hh,
                     const float* __restrict__ proj_w, const float* __restrict__ proj_b,
                     const float* __restrict__ codebook,
                     float* __restrict__ out,
                     float* __restrict__ ws_vqp, int* __restrict__ ws_idx)
{
    extern __shared__ char smem[];
    _Float16* WT0c = (_Float16*)(smem + OFF_WT0C);
    _Float16* WT1c = (_Float16*)(smem + OFF_WT1C);
    _Float16* WT0v = (_Float16*)(smem + OFF_WT0V);
    _Float16* WT1v = (_Float16*)(smem + OFF_WT1V);
    _Float16* A0c  = (_Float16*)(smem + OFF_A0C);
    _Float16* A1c  = (_Float16*)(smem + OFF_A1C);
    _Float16* A0v  = (_Float16*)(smem + OFF_A0V);
    _Float16* A1v  = (_Float16*)(smem + OFF_A1V);
    float* bias0c  = (float*)(smem + OFF_B0C);
    float* bias1c  = (float*)(smem + OFF_B1C);
    float* bias0v  = (float*)(smem + OFF_B0V);
    float* bias1v  = (float*)(smem + OFF_B1V);
    float* convw   = (float*)(smem + OFF_CONVW);
    float* vqproj  = (float*)(smem + OFF_VQP);

    const int tid  = threadIdx.x;
    const int lane = tid & 31;
    const int wv   = tid >> 5;
    const int mb   = wv & 1;     // M-tile (batch rows)
    const int jt   = wv >> 1;    // hidden-unit 16-tile
    const int b0   = blockIdx.x * BT;

    // ---- one-time setup: pack weights transposed+stacked into LDS as f16 ----
    for (int e = tid; e < 256 * 96; e += NTHREADS) {
        int n = e / 96, k = e % 96;
        WT0c[e] = (_Float16)(k < 32 ? w0c_ih[n * 32 + k] : w0c_hh[n * 64 + (k - 32)]);
        float vv = 0.f;
        if (k < 32) { if (k < DIN) vv = w0v_ih[n * DIN + k]; }
        else        { vv = w0v_hh[n * 64 + (k - 32)]; }
        WT0v[e] = (_Float16)vv;
    }
    for (int e = tid; e < 256 * 128; e += NTHREADS) {
        int n = e / 128, k = e % 128;
        WT1c[e] = (_Float16)(k < 64 ? w1c_ih[n * 64 + k] : w1c_hh[n * 64 + (k - 64)]);
        WT1v[e] = (_Float16)(k < 64 ? w1v_ih[n * 64 + k] : w1v_hh[n * 64 + (k - 64)]);
    }
    for (int n = tid; n < 256; n += NTHREADS) {
        bias0c[n] = b0c_ih[n] + b0c_hh[n];
        bias1c[n] = b1c_ih[n] + b1c_hh[n];
        bias0v[n] = b0v_ih[n] + b0v_hh[n];
        bias1v[n] = b1v_ih[n] + b1v_hh[n];
    }
    // conv weights with BN folded: layout [c][r*5+d], slot 15 = fused bias
    for (int e = tid; e < CCONV * 16; e += NTHREADS) {
        int c = e >> 4, k = e & 15;
        float sc = bn_g[c] * rsqrtf(bn_v[c] + 1e-5f);
        float vv;
        if (k < 15) { int r = k / 5, d = k % 5; vv = conv_w[c * 15 + d * 3 + r] * sc; }
        else        { vv = (conv_b[c] - bn_m[c]) * sc + bn_b[c]; }
        convw[e] = vv;
    }
    for (int e = tid; e < BT * 96; e += NTHREADS)  { A0c[e] = (_Float16)0.f; A0v[e] = (_Float16)0.f; }
    for (int e = tid; e < BT * 128; e += NTHREADS) { A1c[e] = (_Float16)0.f; A1v[e] = (_Float16)0.f; }

    v8f c0c = {0,0,0,0,0,0,0,0}, c1c = {0,0,0,0,0,0,0,0};
    v8f c0v = {0,0,0,0,0,0,0,0}, c1v = {0,0,0,0,0,0,0,0};
    float h0c[8], h0v[8], h1c[8], h1v[8];

    __syncthreads();

    // ---- recurrent time loop; weights & activations stay in LDS/registers ----
    for (int t = 0; t < SEQ; ++t) {
        // stage conv(x)+BN+ReLU tile and raw x tile (pad DIN->32 kept zero)
        {
            const int b  = tid >> 3;
            const int cb = (tid & 7) * 4;
            float xv[15];
            const float* xp = x + (size_t)(b0 + b) * SEQ * DIN;
#pragma unroll
            for (int r = 0; r < 3; ++r) {
                const int tt = t - 1 + r;
#pragma unroll
                for (int d = 0; d < DIN; ++d)
                    xv[r * 5 + d] = (tt >= 0 && tt < SEQ) ? xp[tt * DIN + d] : 0.f;
            }
#pragma unroll
            for (int q = 0; q < 4; ++q) {
                const int c = cb + q;
                const float* wr = convw + c * 16;
                float a = wr[15];
#pragma unroll
                for (int k = 0; k < 15; ++k) a = fmaf(xv[k], wr[k], a);
                A0c[b * 96 + c] = (_Float16)fmaxf(a, 0.f);
            }
            if (tid < BT) {
                const float* xq = x + ((size_t)(b0 + tid) * SEQ + t) * DIN;
#pragma unroll
                for (int d = 0; d < DIN; ++d) A0v[tid * 96 + d] = (_Float16)xq[d];
            }
        }
        __syncthreads();   // staging + prev h visible
        lstm_step<96>(A0c, WT0c, bias0c, lane, mb, jt, c0c, h0c);
        lstm_step<96>(A0v, WT0v, bias0v, lane, mb, jt, c0v, h0v);
        __syncthreads();   // all reads of h0_prev done
        write_h(A0c, 96, 32, lane, mb, jt, h0c);   // recurrent input for layer0
        write_h(A1c, 128, 0, lane, mb, jt, h0c);   // feed-forward input for layer1
        write_h(A0v, 96, 32, lane, mb, jt, h0v);
        write_h(A1v, 128, 0, lane, mb, jt, h0v);
        __syncthreads();   // h0 visible
        lstm_step<128>(A1c, WT1c, bias1c, lane, mb, jt, c1c, h1c);
        lstm_step<128>(A1v, WT1v, bias1v, lane, mb, jt, c1v, h1v);
        __syncthreads();   // all reads of h1_prev done
        write_h(A1c, 128, 64, lane, mb, jt, h1c);
        write_h(A1v, 128, 64, lane, mb, jt, h1v);
    }
    __syncthreads();

    // ---- projection: vq_proj = h1v @ proj_w^T + proj_b ----
    {
        const int b  = tid >> 3;
        const int cb = (tid & 7) * 4;
#pragma unroll
        for (int q = 0; q < 4; ++q) {
            const int col = cb + q;
            float a = proj_b[col];
            for (int k = 0; k < HID; ++k)
                a = fmaf((float)A1v[b * 128 + 64 + k], proj_w[col * HID + k], a);
            vqproj[b * EDIM + col] = a;
            ws_vqp[(size_t)(b0 + b) * EDIM + col] = a;
        }
    }
    __syncthreads();

    // ---- VQ argmin + write combined output (both copies) ----
    if (tid < BT) {
        const int b = tid;
        float best = 3.4e38f; int bi = 0;
        for (int n = 0; n < NEMB; ++n) {
            float d = 0.f;
            for (int e = 0; e < EDIM; ++e) {
                float df = vqproj[b * EDIM + e] - codebook[n * EDIM + e];
                d = fmaf(df, df, d);
            }
            if (d < best) { best = d; bi = n; }
        }
        ws_idx[b0 + b] = bi;
        float* o0 = out + (size_t)(b0 + b) * 96;
        float* o1 = out + (size_t)B_TOT * 96 + (size_t)(b0 + b) * 96;
        for (int k = 0; k < HID; ++k) {
            float hv = (float)A1c[b * 128 + 64 + k];  // cnn_features
            o0[k] = hv; o1[k] = hv;
        }
        for (int e = 0; e < EDIM; ++e) {
            float qv = codebook[bi * EDIM + e];       // quantized (STE forward value)
            o0[64 + e] = qv; o1[64 + e] = qv;
        }
    }
}

__global__ __launch_bounds__(NTHREADS)
void vq_loss_kernel(const float* __restrict__ vqp, const int* __restrict__ idx,
                    const float* __restrict__ codebook, float* __restrict__ out)
{
    __shared__ float red[NTHREADS];
    __shared__ int cnt[NEMB];
    const int tid = threadIdx.x;
    if (tid < NEMB) cnt[tid] = 0;
    __syncthreads();
    float local = 0.f;
    for (int e = tid; e < B_TOT * EDIM; e += NTHREADS) {
        const int b = e >> 5, k = e & 31;
        float d = vqp[e] - codebook[idx[b] * EDIM + k];
        local = fmaf(d, d, local);
    }
    for (int b = tid; b < B_TOT; b += NTHREADS) atomicAdd(&cnt[idx[b]], 1);
    red[tid] = local;
    __syncthreads();
    for (int s = NTHREADS / 2; s > 0; s >>= 1) {
        if (tid < s) red[tid] += red[tid + s];
        __syncthreads();
    }
    if (tid == 0) {
        const float mse = red[0] / (float)(B_TOT * EDIM);
        out[2 * B_TOT * 96] = mse * 1.01f;            // q_loss + COMMIT*e_loss (equal numerically)
        float s = 0.f;
        for (int n = 0; n < NEMB; ++n) {
            float p = (float)cnt[n] / (float)B_TOT;
            s += p * logf(p + 1e-10f);
        }
        out[2 * B_TOT * 96 + 1] = __expf(-s);
    }
}

extern "C" void kernel_launch(void* const* d_in, const int* in_sizes, int n_in,
                              void* d_out, int out_size, void* d_ws, size_t ws_size,
                              hipStream_t stream)
{
    const float* x      = (const float*)d_in[0];
    const float* conv_w = (const float*)d_in[1];
    const float* conv_b = (const float*)d_in[2];
    const float* bn_g   = (const float*)d_in[3];
    const float* bn_b   = (const float*)d_in[4];
    const float* bn_m   = (const float*)d_in[5];
    const float* bn_v   = (const float*)d_in[6];
    const float* w0c_ih = (const float*)d_in[7];
    const float* w0c_hh = (const float*)d_in[8];
    const float* b0c_ih = (const float*)d_in[9];
    const float* b0c_hh = (const float*)d_in[10];
    const float* w1c_ih = (const float*)d_in[11];
    const float* w1c_hh = (const float*)d_in[12];
    const float* b1c_ih = (const float*)d_in[13];
    const float* b1c_hh = (const float*)d_in[14];
    const float* w0v_ih = (const float*)d_in[15];
    const float* w0v_hh = (const float*)d_in[16];
    const float* b0v_ih = (const float*)d_in[17];
    const float* b0v_hh = (const float*)d_in[18];
    const float* w1v_ih = (const float*)d_in[19];
    const float* w1v_hh = (const float*)d_in[20];
    const float* b1v_ih = (const float*)d_in[21];
    const float* b1v_hh = (const float*)d_in[22];
    const float* proj_w = (const float*)d_in[23];
    const float* proj_b = (const float*)d_in[24];
    const float* cbook  = (const float*)d_in[25];

    float* ws_vqp = (float*)d_ws;
    int*   ws_idx = (int*)((char*)d_ws + (size_t)B_TOT * EDIM * sizeof(float));

    dim3 grid(B_TOT / BT);
    dual_lstm_fused<<<grid, NTHREADS, SMEM_BYTES, stream>>>(
        x, conv_w, conv_b, bn_g, bn_b, bn_m, bn_v,
        w0c_ih, w0c_hh, b0c_ih, b0c_hh,
        w1c_ih, w1c_hh, b1c_ih, b1c_hh,
        w0v_ih, w0v_hh, b0v_ih, b0v_hh,
        w1v_ih, w1v_hh, b1v_ih, b1v_hh,
        proj_w, proj_b, cbook,
        (float*)d_out, ws_vqp, ws_idx);

    vq_loss_kernel<<<1, NTHREADS, 0, stream>>>(ws_vqp, ws_idx, cbook, (float*)d_out);
}
